// PerResidueEncoder_17987323036197
// MI455X (gfx1250) — matrix operand order repaired and checked by reading, not compiled, not run on hardware
//
#include <hip/hip_runtime.h>
#include <hip/hip_bf16.h>

typedef _Float16 half16 __attribute__((ext_vector_type(16)));
typedef _Float16 half8  __attribute__((ext_vector_type(8)));
typedef float    float8 __attribute__((ext_vector_type(8)));
typedef int      v4i_t  __attribute__((ext_vector_type(4)));

#if defined(__has_builtin)
#if __has_builtin(__builtin_amdgcn_global_load_async_to_lds_b128) && \
    __has_builtin(__builtin_amdgcn_global_load_async_to_lds_b32)
#define HAVE_ASYNC_LDS 1
#endif
#endif
#ifndef HAVE_ASYNC_LDS
#define HAVE_ASYNC_LDS 0
#endif

// ---------------------------------------------------------------------------
// Weight swizzle: f32 row-major W[K][N] -> f16 packed WMMA-B fragment layout.
// Bp[((kt*N + n)*2 + half)*16 + e] = W[kt*32 + half*8 + (e&7) + (e>>3)*16][n]
// (B 32x16 f16 layout: lanes 0-15 = cols with K 0..7,16..23, lanes 16-31 = K 8..15,24..31)
// ---------------------------------------------------------------------------
__global__ void swizzleB(const float* __restrict__ W, _Float16* __restrict__ Bp,
                         int K, int Kpad, int N) {
    int i = blockIdx.x * blockDim.x + threadIdx.x;
    int total = (Kpad >> 5) * N * 32;
    if (i >= total) return;
    int e    = i & 15;
    int half = (i >> 4) & 1;
    int n    = (i >> 5) % N;
    int kt   = (i >> 5) / N;
    int k    = kt * 32 + half * 8 + (e & 7) + (e >> 3) * 16;
    Bp[i] = (k < K) ? (_Float16)W[(size_t)k * N + n] : (_Float16)0.f;
}

// ---------------------------------------------------------------------------
// E[a][j] = b1[j] + sum_k embed[a][k] * w1[882+k][j]   (21 x 1024, f32)
// ---------------------------------------------------------------------------
__global__ void embed_proj(const float* __restrict__ embed, const float* __restrict__ w1,
                           const float* __restrict__ b1, float* __restrict__ E) {
    int i = blockIdx.x * blockDim.x + threadIdx.x;
    if (i >= 21 * 1024) return;
    int a = i >> 10, j = i & 1023;
    float s = b1[j];
    for (int k = 0; k < 512; ++k)
        s += embed[a * 512 + k] * w1[(size_t)(882 + k) * 1024 + j];
    E[i] = s;
}

// ---------------------------------------------------------------------------
// Geometry: local frame + rotate 14 atoms, masked; store f16 crd[M][48]
// ---------------------------------------------------------------------------
__global__ void geom_kernel(const float* __restrict__ pos,
                            const unsigned char* __restrict__ mask,
                            _Float16* __restrict__ crd, int M) {
    int r = blockIdx.x * blockDim.x + threadIdx.x;
    if (r >= M) return;
    const float* p = pos + (size_t)r * 42;
    float cx = p[3], cy = p[4], cz = p[5];
    float v1x = p[6] - cx, v1y = p[7] - cy, v1z = p[8] - cz;
    float n1 = sqrtf(v1x * v1x + v1y * v1y + v1z * v1z) + 1e-6f;
    float e1x = v1x / n1, e1y = v1y / n1, e1z = v1z / n1;
    float v2x = p[0] - cx, v2y = p[1] - cy, v2z = p[2] - cz;
    float d = e1x * v2x + e1y * v2y + e1z * v2z;
    float u2x = v2x - d * e1x, u2y = v2y - d * e1y, u2z = v2z - d * e1z;
    float n2 = sqrtf(u2x * u2x + u2y * u2y + u2z * u2z) + 1e-6f;
    float e2x = u2x / n2, e2y = u2y / n2, e2z = u2z / n2;
    float e3x = e1y * e2z - e1z * e2y;
    float e3y = e1z * e2x - e1x * e2z;
    float e3z = e1x * e2y - e1y * e2x;
    _Float16* o = crd + (size_t)r * 48;
#pragma unroll
    for (int a = 0; a < 14; ++a) {
        float qx = p[a * 3 + 0] - cx, qy = p[a * 3 + 1] - cy, qz = p[a * 3 + 2] - cz;
        float m = mask[(size_t)r * 14 + a] ? 1.f : 0.f;
        o[a * 3 + 0] = (_Float16)((qx * e1x + qy * e1y + qz * e1z) * m);
        o[a * 3 + 1] = (_Float16)((qx * e2x + qy * e2y + qz * e2z) * m);
        o[a * 3 + 2] = (_Float16)((qx * e3x + qy * e3y + qz * e3z) * m);
    }
#pragma unroll
    for (int t = 42; t < 48; ++t) o[t] = (_Float16)0.f;
}

// ---------------------------------------------------------------------------
// Layer 1: h1 = relu(onehot-sparse crd GEMM + E[aa]) ; K=896 (padded 882), N=1024
// crd rows staged in LDS (async global->LDS on gfx1250) with 32-half zero guard
// bands; A fragments are two contiguous 8-half runs read branchlessly from the
// guarded window. LDS row stride = 136 halfs (68 dwords) -> 4-bank rotation.
// ---------------------------------------------------------------------------
#define LROW 136
__global__ __launch_bounds__(256) void gemm_layer1(
    const _Float16* __restrict__ crd, const int* __restrict__ aa,
    const _Float16* __restrict__ Bp, const float* __restrict__ E,
    _Float16* __restrict__ h1, int M) {
    const int N = 1024;
    __shared__ _Float16 lcrd[128][LROW];  // ~34 KB

    int tid  = threadIdx.x;
    int lane = tid & 31, wid = tid >> 5;
    int wave_r = wid & 3, wave_c = wid >> 2;
    int blockRow0 = blockIdx.y * 128;
    int row0 = blockRow0 + wave_r * 32;
    int colw = blockIdx.x * 128 + wave_c * 64;
    int lrow = lane & 15, lhalf = lane >> 4;

    // ---- zero guard bands: dwords [0,16) and [37,68) of each row ----
    {
        int row_l = tid >> 1, part = tid & 1;
        uint32_t* z = (uint32_t*)&lcrd[row_l][0];
        if (part == 0) {
#pragma unroll
            for (int i = 0; i < 16; ++i) z[i] = 0;
            z[37] = 0;
        } else {
#pragma unroll
            for (int i = 38; i < 68; ++i) z[i] = 0;
        }
    }
    // ---- copy 42 crd halfs (84 B) into dwords [16,37) of each row ----
    if (tid < 128) {
        const char* src = (const char*)(crd + (size_t)(blockRow0 + tid) * 48);
        char*       dst = (char*)&lcrd[tid][32];
#if HAVE_ASYNC_LDS
        typedef __attribute__((address_space(1))) char  gchar;
        typedef __attribute__((address_space(3))) char  lchar;
        typedef __attribute__((address_space(1))) v4i_t gv4;
        typedef __attribute__((address_space(3))) v4i_t lv4;
        typedef __attribute__((address_space(1))) int   gi32;
        typedef __attribute__((address_space(3))) int   li32;
        gchar* g = (gchar*)src;
        lchar* l = (lchar*)dst;
#pragma unroll
        for (int i = 0; i < 5; ++i)
            __builtin_amdgcn_global_load_async_to_lds_b128(
                (gv4*)(g + i * 16), (lv4*)(l + i * 16), 0, 0);
        __builtin_amdgcn_global_load_async_to_lds_b32(
            (gi32*)(g + 80), (li32*)(l + 80), 0, 0);
#else
#pragma unroll
        for (int i = 0; i < 5; ++i)
            ((float4*)dst)[i] = ((const float4*)src)[i];
        ((uint32_t*)dst)[20] = ((const uint32_t*)src)[20];
#endif
    }
#if HAVE_ASYNC_LDS
    asm volatile("s_wait_asynccnt 0x0" ::: "memory");
#endif
    __syncthreads();

    int rowL[2], sa[2];
#pragma unroll
    for (int s = 0; s < 2; ++s) {
        rowL[s] = wave_r * 32 + s * 16 + lrow;
        sa[s]   = aa[blockRow0 + rowL[s]] * 42;
    }
    float8 acc[2][4] = {};

    for (int kt = 0; kt < 28; ++kt) {
        int k0 = kt * 32;
        bool p = false;
#pragma unroll
        for (int s = 0; s < 2; ++s)
            p = p || ((sa[s] < k0 + 32) && (sa[s] + 42 > k0));
        if (__ballot(p) == 0ull) continue;  // wave-uniform skip, EXEC stays all-1

        if (kt + 1 < 28)
            __builtin_prefetch(
                Bp + ((size_t)((kt + 1) * N + colw + lrow) * 2 + lhalf) * 16, 0, 1);

        half16 a[2];
#pragma unroll
        for (int s = 0; s < 2; ++s) {
            const uint32_t* rp = (const uint32_t*)&lcrd[rowL[s]][0];
            int kb = k0 + lhalf * 8 - sa[s];  // always even
            union { half16 v; uint32_t u[8]; } av;
#pragma unroll
            for (int r = 0; r < 2; ++r) {
                int  base = kb + r * 16;
                bool zero = (base <= -8) || (base >= 42);
                int  sb   = zero ? 0 : base;     // in [-7,41] when live
                int  di   = (32 + sb) >> 1;      // dword index into guarded row
#pragma unroll
                for (int q = 0; q < 4; ++q) {
                    uint32_t x = rp[di + q];
                    av.u[r * 4 + q] = zero ? 0u : x;
                }
            }
            a[s] = av.v;
        }
        half16 b[4];
#pragma unroll
        for (int c = 0; c < 4; ++c) {
            const _Float16* bp =
                Bp + ((size_t)(kt * N + (colw + c * 16 + lrow)) * 2 + lhalf) * 16;
            b[c] = *(const half16*)bp;
        }
#pragma unroll
        for (int s = 0; s < 2; ++s)
#pragma unroll
            for (int c = 0; c < 4; ++c)
                acc[s][c] = __builtin_amdgcn_wmma_f32_16x16x32_f16(
                    false, a[s], false, b[c], (short)0, acc[s][c], false, false);
    }

    // epilogue: + E[aa[row]][col], relu, store f16
    int ar[2][8];
#pragma unroll
    for (int s = 0; s < 2; ++s)
#pragma unroll
        for (int j = 0; j < 8; ++j)
            ar[s][j] = aa[row0 + s * 16 + lhalf * 8 + j];
#pragma unroll
    for (int s = 0; s < 2; ++s)
#pragma unroll
        for (int c = 0; c < 4; ++c) {
            int col = colw + c * 16 + lrow;
#pragma unroll
            for (int j = 0; j < 8; ++j) {
                int row = row0 + s * 16 + lhalf * 8 + j;
                float v = acc[s][c][j] + E[(size_t)ar[s][j] * 1024 + col];
                v = fmaxf(v, 0.f);
                h1[(size_t)row * N + col] = (_Float16)v;
            }
        }
    (void)M;
}

// ---------------------------------------------------------------------------
// Generic fused GEMM: out = [relu](A[M,K]f16 @ Wp + bias); out f16 or f32
// 128x128 block tile, 8 waves, each wave 32 rows x 64 cols (2x4 WMMA accs).
// ---------------------------------------------------------------------------
template <bool RELU, bool OUTF16>
__global__ __launch_bounds__(256) void gemm_mlp(
    const _Float16* __restrict__ A, const _Float16* __restrict__ Bp,
    const float* __restrict__ bias, void* __restrict__ out, int K, int N) {
    int lane = threadIdx.x & 31, wid = threadIdx.x >> 5;
    int wave_r = wid & 3, wave_c = wid >> 2;
    int row0 = blockIdx.y * 128 + wave_r * 32;
    int colw = blockIdx.x * 128 + wave_c * 64;
    int lrow = lane & 15, lhalf = lane >> 4;

    float8 acc[2][4] = {};
    int ktiles = K >> 5;
    for (int kt = 0; kt < ktiles; ++kt) {
        if (kt + 1 < ktiles)  // gfx1250 global_prefetch of next B K-tile
            __builtin_prefetch(
                Bp + ((size_t)((kt + 1) * N + colw + lrow) * 2 + lhalf) * 16, 0, 1);
        half16 a[2], b[4];
#pragma unroll
        for (int s = 0; s < 2; ++s) {
            const _Float16* ap =
                A + (size_t)(row0 + s * 16 + lrow) * K + kt * 32 + lhalf * 8;
            half8 lo = *(const half8*)ap;
            half8 hi = *(const half8*)(ap + 16);
            a[s] = __builtin_shufflevector(lo, hi, 0, 1, 2, 3, 4, 5, 6, 7, 8, 9,
                                           10, 11, 12, 13, 14, 15);
        }
#pragma unroll
        for (int c = 0; c < 4; ++c) {
            const _Float16* bp =
                Bp + ((size_t)(kt * N + (colw + c * 16 + lrow)) * 2 + lhalf) * 16;
            b[c] = *(const half16*)bp;
        }
#pragma unroll
        for (int s = 0; s < 2; ++s)
#pragma unroll
            for (int c = 0; c < 4; ++c)
                acc[s][c] = __builtin_amdgcn_wmma_f32_16x16x32_f16(
                    false, a[s], false, b[c], (short)0, acc[s][c], false, false);
    }
#pragma unroll
    for (int s = 0; s < 2; ++s)
#pragma unroll
        for (int c = 0; c < 4; ++c) {
            int col = colw + c * 16 + lrow;
            float bv = bias[col];
#pragma unroll
            for (int j = 0; j < 8; ++j) {
                int row = row0 + s * 16 + lhalf * 8 + j;
                float v = acc[s][c][j] + bv;
                if (RELU) v = fmaxf(v, 0.f);
                if (OUTF16)
                    ((_Float16*)out)[(size_t)row * N + col] = (_Float16)v;
                else
                    ((float*)out)[(size_t)row * N + col] = v;
            }
        }
}

// ---------------------------------------------------------------------------
extern "C" void kernel_launch(void* const* d_in, const int* in_sizes, int n_in,
                              void* d_out, int out_size, void* d_ws, size_t ws_size,
                              hipStream_t stream) {
    const int*           aa    = (const int*)d_in[0];
    const float*         pos14 = (const float*)d_in[1];
    const unsigned char* amask = (const unsigned char*)d_in[2];
    const float*         embed = (const float*)d_in[3];
    const float*         w1    = (const float*)d_in[4];
    const float*         b1    = (const float*)d_in[5];
    const float*         w2    = (const float*)d_in[6];
    const float*         b2    = (const float*)d_in[7];
    const float*         w3    = (const float*)d_in[8];
    const float*         b3    = (const float*)d_in[9];
    const float*         w4    = (const float*)d_in[10];
    const float*         b4    = (const float*)d_in[11];
    (void)n_in; (void)out_size; (void)ws_size;

    const int M = in_sizes[0];  // N*L = 65536

    char* ws = (char*)d_ws;
    size_t off = 0;
    auto take = [&](size_t bytes) {
        char* p = ws + off;
        off = (off + bytes + 255) & ~(size_t)255;
        return p;
    };
    _Float16* crd = (_Float16*)take((size_t)M * 48 * 2);
    _Float16* w1p = (_Float16*)take((size_t)896 * 1024 * 2);
    _Float16* w2p = (_Float16*)take((size_t)1024 * 512 * 2);
    _Float16* w3p = (_Float16*)take((size_t)512 * 512 * 2);
    _Float16* w4p = (_Float16*)take((size_t)512 * 512 * 2);
    float*    E   = (float*)take((size_t)21 * 1024 * 4);
    _Float16* h1  = (_Float16*)take((size_t)M * 1024 * 2);
    _Float16* h2  = (_Float16*)take((size_t)M * 512 * 2);
    _Float16* h3  = (_Float16*)take((size_t)M * 512 * 2);

    {
        int tot = 896 * 1024;
        swizzleB<<<(tot + 255) / 256, 256, 0, stream>>>(w1, w1p, 882, 896, 1024);
    }
    {
        int tot = 1024 * 512;
        swizzleB<<<(tot + 255) / 256, 256, 0, stream>>>(w2, w2p, 1024, 1024, 512);
    }
    {
        int tot = 512 * 512;
        swizzleB<<<(tot + 255) / 256, 256, 0, stream>>>(w3, w3p, 512, 512, 512);
        swizzleB<<<(tot + 255) / 256, 256, 0, stream>>>(w4, w4p, 512, 512, 512);
    }
    embed_proj<<<(21 * 1024 + 255) / 256, 256, 0, stream>>>(embed, w1, b1, E);
    geom_kernel<<<(M + 255) / 256, 256, 0, stream>>>(pos14, amask, crd, M);

    gemm_layer1<<<dim3(1024 / 128, M / 128), 256, 0, stream>>>(crd, aa, w1p, E, h1, M);
    gemm_mlp<true, true><<<dim3(512 / 128, M / 128), 256, 0, stream>>>(h1, w2p, b2, h2, 1024, 512);
    gemm_mlp<true, true><<<dim3(512 / 128, M / 128), 256, 0, stream>>>(h2, w3p, b3, h3, 512, 512);
    gemm_mlp<false, false><<<dim3(512 / 128, M / 128), 256, 0, stream>>>(h3, w4p, b4, d_out, 512, 512);
}
